// CriticNetAblation_74543452389449
// MI455X (gfx1250) — compile-verified, matrix operand order checked by reading, not run on hardware
//
#include <hip/hip_runtime.h>
#include <hip/hip_bf16.h>

// ---------------- types ----------------
typedef __attribute__((ext_vector_type(16))) __bf16 v16bf;
typedef __attribute__((ext_vector_type(8)))  float  v8f;
typedef __attribute__((ext_vector_type(4)))  unsigned int v4u;

union Frag {
  v4u            u[2];
  unsigned short s[16];
  v16bf          v;
};

#define DIM   64
#define NN    10000
#define EE    40000
#define GW    4160              // g row width: 64*64 + 64 bias cols
#define NT_G  260               // GW/16
#define MT    625               // NN/16
#define NTPW  4                 // n-tiles per wave in k_g
#define NGRP  (NT_G / NTPW)     // 65

__device__ __forceinline__ unsigned short f2bf(float f) {
  unsigned u = __float_as_uint(f);
  u += 0x7FFFu + ((u >> 16) & 1u);
  return (unsigned short)(u >> 16);
}
__device__ __forceinline__ float bf2f(unsigned short s) {
  return __uint_as_float(((unsigned)s) << 16);
}
__device__ __forceinline__ float sigm(float x) { return 1.0f / (1.0f + expf(-x)); }

__device__ __forceinline__ v8f wmma_bf16(v16bf a, v16bf b, v8f c) {
  return __builtin_amdgcn_wmma_f32_16x16x32_bf16(false, a, false, b, (short)0, c, false, false);
}

// B fragment fetch: swizzled layout [tile][kstep][lane][16 bf16] (32B contiguous per lane)
__device__ __forceinline__ v16bf ldB(const unsigned short* base, int tile, int ks, int lane) {
  return *(const v16bf*)(base + ((((tile * 2 + ks) * 32) + lane) << 4));
}
// A fragment from row-major bf16 [rows][64]: two contiguous 8-element K runs per lane
__device__ __forceinline__ void ldA(Frag& f, const unsigned short* row, int ks, int half) {
  f.u[0] = *(const v4u*)(row + ks * 32 + 8 * half);
  f.u[1] = *(const v4u*)(row + ks * 32 + 16 + 8 * half);
}

__device__ void atomicMaxF(float* addr, float val) {
  int* ia = (int*)addr;
  int old = *ia;
  while (__int_as_float(old) < val) {
    int assumed = old;
    old = atomicCAS(ia, assumed, __float_as_int(val));
    if (old == assumed) break;
  }
}

// ---------------- weight swizzle (once per call) ----------------
// Segments: B2 (260 tiles), root (4), wih (12), whh (12); 1024 elems per tile.
__global__ __launch_bounds__(256) void k_prep(
    const float* __restrict__ em2_w, const float* __restrict__ em2_b,
    const float* __restrict__ root_w, const float* __restrict__ wih,
    const float* __restrict__ whh,
    unsigned short* B2sw, unsigned short* Brootsw,
    unsigned short* Bwihsw, unsigned short* Bwhhsw) {
  int id = blockIdx.x * 256 + threadIdx.x;
  const int C0 = NT_G * 1024, C1 = 4 * 1024, C2 = 12 * 1024;
  int seg, idx;
  if      (id < C0)           { seg = 0; idx = id; }
  else if (id < C0 + C1)      { seg = 1; idx = id - C0; }
  else if (id < C0 + C1 + C2) { seg = 2; idx = id - C0 - C1; }
  else                        { seg = 3; idx = id - C0 - C1 - C2; }
  int q    = idx & 15;
  int lane = (idx >> 4) & 31;
  int ks   = (idx >> 9) & 1;
  int tile = idx >> 10;
  int half = lane >> 4;
  int K    = ks * 32 + (q < 8 ? 8 * half + q : 16 + 8 * half + (q - 8));
  int col  = tile * 16 + (lane & 15);
  float v;
  if (seg == 0) {
    if (col < DIM * DIM) { int o = col >> 6, kk = col & 63; v = em2_w[(K * 64 + o) * 64 + kk]; }
    else                 { v = em2_b[K * 64 + (col - DIM * DIM)]; }
    B2sw[idx] = f2bf(v);
  } else if (seg == 1) {
    v = root_w[K * 64 + col];           Brootsw[idx] = f2bf(v);
  } else if (seg == 2) {
    v = wih[col * 64 + K];              Bwihsw[idx] = f2bf(v);
  } else {
    v = whh[col * 64 + K];              Bwhhsw[idx] = f2bf(v);
  }
}

// ---------------- degree / lin0 / edge MLP stage 1 ----------------
__global__ __launch_bounds__(256) void k_deg(const int* __restrict__ dst, float* deg) {
  int e = blockIdx.x * 256 + threadIdx.x;
  if (e < EE) atomicAdd(&deg[dst[e]], 1.0f);
}

__global__ __launch_bounds__(256) void k_lin0(
    const float* __restrict__ x, const float* __restrict__ w, const float* __restrict__ b,
    const float* __restrict__ deg, float* hf32, unsigned short* hb16, float* invdeg) {
  int id = blockIdx.x * 256 + threadIdx.x;
  if (id >= NN * DIM) return;
  int n = id >> 6, d = id & 63;
  float v = b[d] + x[n * 3 + 0] * w[d * 3 + 0] + x[n * 3 + 1] * w[d * 3 + 1] +
            x[n * 3 + 2] * w[d * 3 + 2];
  v = fmaxf(v, 0.0f);
  hf32[id] = v;
  hb16[id] = f2bf(v);
  if (d == 0) { float dg = deg[n]; invdeg[n] = dg > 0.0f ? 1.0f / dg : 0.0f; }
}

__global__ __launch_bounds__(256) void k_em1(
    const float* __restrict__ ea, const float* __restrict__ w, const float* __restrict__ b,
    unsigned short* h1) {
  int id = blockIdx.x * 256 + threadIdx.x;
  if (id >= EE * DIM) return;
  int e = id >> 6, d = id & 63;
  float v = b[d];
#pragma unroll
  for (int c = 0; c < 7; ++c) v += ea[e * 7 + c] * w[d * 7 + c];
  h1[id] = f2bf(fmaxf(v, 0.0f));
}

// ---------------- g = h @ B2  (WMMA bf16 GEMM, N x 4160) ----------------
// One wave: one 16-row m-tile x NTPW consecutive n-tiles. A fragments loaded
// once; 2*NTPW WMMAs with independent accumulators hide the WMMA RAW latency.
__global__ __launch_bounds__(256) void k_g(
    const unsigned short* __restrict__ hb16, const unsigned short* __restrict__ B2sw,
    unsigned short* __restrict__ g) {
  int wave = threadIdx.x >> 5, lane = threadIdx.x & 31;
  int wid = blockIdx.x * 8 + wave;
  if (wid >= MT * NGRP) return;
  int mtile = wid / NGRP;
  int nt0   = (wid % NGRP) * NTPW;
  int half = lane >> 4, mr = lane & 15;
  const unsigned short* arow = hb16 + (size_t)(mtile * 16 + mr) * DIM;
  Frag a0, a1;
  ldA(a0, arow, 0, half);
  ldA(a1, arow, 1, half);
  size_t rowbase = (size_t)(mtile * 16) * GW;
#pragma unroll
  for (int tt = 0; tt < NTPW; ++tt) {
    int ntile = nt0 + tt;
    v8f c = {};
    c = wmma_bf16(a0.v, ldB(B2sw, ntile, 0, lane), c);
    c = wmma_bf16(a1.v, ldB(B2sw, ntile, 1, lane), c);
    int ncol = ntile * 16 + mr;
#pragma unroll
    for (int r = 0; r < 8; ++r) {
      g[rowbase + (size_t)(r + 8 * half) * GW + ncol] = f2bf(c[r]);
    }
  }
}

// ---------------- messages + scatter (bandwidth kernel, L2-resident g) ------
__global__ __launch_bounds__(256) void k_msg(
    const unsigned short* __restrict__ h1, const unsigned short* __restrict__ g,
    const int* __restrict__ src, const int* __restrict__ dst, float* agg) {
  __shared__ float h1s[4][DIM];
  int le = threadIdx.x >> 6;           // edge slot in block
  int o  = threadIdx.x & 63;
  int e  = blockIdx.x * 4 + le;        // EE/4 blocks exactly
  int s = src[e], d = dst[e];
  const unsigned short* gp = g + (size_t)s * GW + o * 64;
  __builtin_prefetch(gp, 0, 1);        // global_prefetch_b8: pull row while we stage h1
  __builtin_prefetch(gp + 32, 0, 1);
  h1s[le][o] = bf2f(h1[(size_t)e * DIM + o]);
  __syncthreads();
  const v4u* gv = (const v4u*)gp;
  float acc = 0.0f;
#pragma unroll
  for (int blk = 0; blk < 8; ++blk) {
    v4u wv = gv[blk];
    const float* hh = &h1s[le][blk * 8];
    acc += hh[0] * __uint_as_float(wv.x << 16);
    acc += hh[1] * __uint_as_float(wv.x & 0xFFFF0000u);
    acc += hh[2] * __uint_as_float(wv.y << 16);
    acc += hh[3] * __uint_as_float(wv.y & 0xFFFF0000u);
    acc += hh[4] * __uint_as_float(wv.z << 16);
    acc += hh[5] * __uint_as_float(wv.z & 0xFFFF0000u);
    acc += hh[6] * __uint_as_float(wv.w << 16);
    acc += hh[7] * __uint_as_float(wv.w & 0xFFFF0000u);
  }
  acc += bf2f(g[(size_t)s * GW + DIM * DIM + o]);   // folded em2_b term
  atomicAdd(&agg[(size_t)d * DIM + o], acc);
}

// ---------------- fused NNConv-root + GRU cell (WMMA) ----------------
__global__ __launch_bounds__(256) void k_gru(
    unsigned short* hb16, float* hf32,
    const float* __restrict__ agg, const float* __restrict__ invdeg,
    const float* __restrict__ conv_b,
    const unsigned short* __restrict__ Broot, const unsigned short* __restrict__ Bwih,
    const unsigned short* __restrict__ Bwhh,
    const float* __restrict__ bih, const float* __restrict__ bhh) {
  __shared__ float msh[8][16 * DIM];
  int wave = threadIdx.x >> 5, lane = threadIdx.x & 31;
  int wid = blockIdx.x * 8 + wave;
  bool active = wid < MT;
  int half = lane >> 4, mr = lane & 15;
  Frag a0, a1;                          // A fragments of h (== out)
  if (active) {
    const unsigned short* arow = hb16 + (size_t)(wid * 16 + mr) * DIM;
    ldA(a0, arow, 0, half);
    ldA(a1, arow, 1, half);
    // m = relu(out @ root_w + agg*inv_deg + conv_b)  -> LDS tile
#pragma unroll
    for (int t = 0; t < 4; ++t) {
      v8f c = {};
      c = wmma_bf16(a0.v, ldB(Broot, t, 0, lane), c);
      c = wmma_bf16(a1.v, ldB(Broot, t, 1, lane), c);
      int col = t * 16 + mr;
#pragma unroll
      for (int r = 0; r < 8; ++r) {
        int row = wid * 16 + r + 8 * half;
        float v = c[r] + agg[(size_t)row * DIM + col] * invdeg[row] + conv_b[col];
        msh[wave][(r + 8 * half) * DIM + col] = fmaxf(v, 0.0f);
      }
    }
  }
  __syncthreads();
  if (active) {
    // re-fragment m from LDS as bf16 A
    Frag am0, am1;
    const float* mrow = &msh[wave][mr * DIM];
#pragma unroll
    for (int q = 0; q < 8; ++q) {
      am0.s[q]     = f2bf(mrow[8 * half + q]);
      am0.s[8 + q] = f2bf(mrow[16 + 8 * half + q]);
      am1.s[q]     = f2bf(mrow[32 + 8 * half + q]);
      am1.s[8 + q] = f2bf(mrow[48 + 8 * half + q]);
    }
#pragma unroll
    for (int t = 0; t < 4; ++t) {
      v8f ci0 = {}, ci1 = {}, ci2 = {}, ch0 = {}, ch1 = {}, ch2 = {};
      ci0 = wmma_bf16(am0.v, ldB(Bwih, t + 0, 0, lane), ci0);
      ci0 = wmma_bf16(am1.v, ldB(Bwih, t + 0, 1, lane), ci0);
      ci1 = wmma_bf16(am0.v, ldB(Bwih, t + 4, 0, lane), ci1);
      ci1 = wmma_bf16(am1.v, ldB(Bwih, t + 4, 1, lane), ci1);
      ci2 = wmma_bf16(am0.v, ldB(Bwih, t + 8, 0, lane), ci2);
      ci2 = wmma_bf16(am1.v, ldB(Bwih, t + 8, 1, lane), ci2);
      ch0 = wmma_bf16(a0.v,  ldB(Bwhh, t + 0, 0, lane), ch0);
      ch0 = wmma_bf16(a1.v,  ldB(Bwhh, t + 0, 1, lane), ch0);
      ch1 = wmma_bf16(a0.v,  ldB(Bwhh, t + 4, 0, lane), ch1);
      ch1 = wmma_bf16(a1.v,  ldB(Bwhh, t + 4, 1, lane), ch1);
      ch2 = wmma_bf16(a0.v,  ldB(Bwhh, t + 8, 0, lane), ch2);
      ch2 = wmma_bf16(a1.v,  ldB(Bwhh, t + 8, 1, lane), ch2);
      int col = t * 16 + mr;
      float br = bih[col],        bz = bih[64 + col],  bn = bih[128 + col];
      float cr = bhh[col],        cz = bhh[64 + col],  cn = bhh[128 + col];
#pragma unroll
      for (int r = 0; r < 8; ++r) {
        int row = wid * 16 + r + 8 * half;
        size_t off = (size_t)row * DIM + col;
        float hold = hf32[off];
        float rg = sigm(ci0[r] + br + ch0[r] + cr);
        float z  = sigm(ci1[r] + bz + ch1[r] + cz);
        float nc = tanhf(ci2[r] + bn + rg * (ch2[r] + cn));
        float hn = (1.0f - z) * nc + z * hold;
        hf32[off] = hn;
        hb16[off] = f2bf(hn);
      }
    }
  }
}

// ---------------- Set2Set ----------------
__global__ __launch_bounds__(256) void k_lstm(
    const float* __restrict__ qstar, float* hl, float* cl, float* q,
    const float* __restrict__ wih, const float* __restrict__ whh,
    const float* __restrict__ bih, const float* __restrict__ bhh,
    float* attnmax, float* attnden, float* rvec) {
  __shared__ float qs[2 * DIM];
  __shared__ float hs[DIM];
  __shared__ float gsh[4 * DIM];
  int tid = threadIdx.x;
  if (tid < 2 * DIM) qs[tid] = qstar[tid];
  if (tid < DIM) hs[tid] = hl[tid];
  __syncthreads();
  float acc = bih[tid] + bhh[tid];
#pragma unroll 8
  for (int k = 0; k < 2 * DIM; ++k) acc += qs[k] * wih[tid * 2 * DIM + k];
#pragma unroll 8
  for (int k = 0; k < DIM; ++k) acc += hs[k] * whh[tid * DIM + k];
  gsh[tid] = acc;
  __syncthreads();
  if (tid < DIM) {
    float gi = gsh[tid], gf = gsh[DIM + tid], gg = gsh[2 * DIM + tid], go = gsh[3 * DIM + tid];
    float c = sigm(gf) * cl[tid] + sigm(gi) * tanhf(gg);
    float h = sigm(go) * tanhf(c);
    cl[tid] = c; hl[tid] = h; q[tid] = h;
    rvec[tid] = 0.0f;
  }
  if (tid == 0) { attnmax[0] = -3.4e38f; attnden[0] = 0.0f; }
}

__global__ __launch_bounds__(256) void k_attn_logits(
    const float* __restrict__ out, const float* __restrict__ q,
    float* evec, float* attnmax) {
  __shared__ float qs[DIM];
  __shared__ float red[256];
  int tid = threadIdx.x;
  int n = blockIdx.x * 256 + tid;
  if (tid < DIM) qs[tid] = q[tid];
  __syncthreads();
  float e = -3.4e38f;
  if (n < NN) {
    float acc = 0.0f;
    const float* row = out + (size_t)n * DIM;
#pragma unroll 8
    for (int d = 0; d < DIM; ++d) acc += row[d] * qs[d];
    evec[n] = acc;
    e = acc;
  }
  red[tid] = e;
  __syncthreads();
  for (int s = 128; s > 0; s >>= 1) {
    if (tid < s) red[tid] = fmaxf(red[tid], red[tid + s]);
    __syncthreads();
  }
  if (tid == 0) atomicMaxF(attnmax, red[0]);
}

__global__ __launch_bounds__(256) void k_attn_den(
    float* evec, const float* __restrict__ attnmax, float* attnden) {
  __shared__ float red[256];
  int tid = threadIdx.x;
  int n = blockIdx.x * 256 + tid;
  float ex = 0.0f;
  float mx = attnmax[0];
  if (n < NN) { ex = expf(evec[n] - mx); evec[n] = ex; }
  red[tid] = ex;
  __syncthreads();
  for (int s = 128; s > 0; s >>= 1) {
    if (tid < s) red[tid] += red[tid + s];
    __syncthreads();
  }
  if (tid == 0) atomicAdd(attnden, red[0]);
}

__global__ __launch_bounds__(256) void k_attn_r(
    const float* __restrict__ evec, const float* __restrict__ out,
    const float* __restrict__ attnden, float* rvec) {
  __shared__ float sh[4][DIM];
  int tid = threadIdx.x;
  int slot = tid >> 6, d = tid & 63;
  float acc = 0.0f;
  for (int jj = slot; jj < 64; jj += 4) {
    int n = blockIdx.x * 64 + jj;
    if (n < NN) acc += evec[n] * out[(size_t)n * DIM + d];
  }
  sh[slot][d] = acc;
  __syncthreads();
  if (slot == 0) {
    float tot = sh[0][d] + sh[1][d] + sh[2][d] + sh[3][d];
    atomicAdd(&rvec[d], tot / attnden[0]);
  }
}

__global__ void k_qstar(const float* __restrict__ q, const float* __restrict__ rvec,
                        float* qstar) {
  int tid = threadIdx.x;
  if (tid < DIM) qstar[tid] = q[tid];
  else if (tid < 2 * DIM) qstar[tid] = rvec[tid - DIM];
}

__global__ void k_final(const float* __restrict__ qstar,
                        const float* __restrict__ w1, const float* __restrict__ b1,
                        const float* __restrict__ w3, const float* __restrict__ b3,
                        float* outp) {
  __shared__ float hid[DIM];
  int tid = threadIdx.x;
  if (tid < DIM) {
    float acc = b1[tid];
#pragma unroll 8
    for (int k = 0; k < 2 * DIM; ++k) acc += qstar[k] * w1[tid * 2 * DIM + k];
    hid[tid] = fmaxf(acc, 0.0f);
  }
  __syncthreads();
  if (tid == 0) {
    float acc = b3[0];
    for (int d = 0; d < DIM; ++d) acc += hid[d] * w3[d];
    outp[0] = acc;
  }
}

// ---------------- host ----------------
static inline size_t alignUp(size_t v, size_t a) { return (v + a - 1) & ~(a - 1); }

extern "C" void kernel_launch(void* const* d_in, const int* in_sizes, int n_in,
                              void* d_out, int out_size, void* d_ws, size_t ws_size,
                              hipStream_t stream) {
  const float* x        = (const float*)d_in[0];
  const float* ea       = (const float*)d_in[1];
  const int*   eidx     = (const int*)  d_in[2];
  // d_in[3] = batch (unused, all zeros)
  const float* lin0_w   = (const float*)d_in[4];
  const float* lin0_b   = (const float*)d_in[5];
  const float* em1_w    = (const float*)d_in[6];
  const float* em1_b    = (const float*)d_in[7];
  const float* em2_w    = (const float*)d_in[8];
  const float* em2_b    = (const float*)d_in[9];
  const float* root_w   = (const float*)d_in[10];
  const float* conv_b   = (const float*)d_in[11];
  const float* gru_wih  = (const float*)d_in[12];
  const float* gru_whh  = (const float*)d_in[13];
  const float* gru_bih  = (const float*)d_in[14];
  const float* gru_bhh  = (const float*)d_in[15];
  const float* lstm_wih = (const float*)d_in[16];
  const float* lstm_whh = (const float*)d_in[17];
  const float* lstm_bih = (const float*)d_in[18];
  const float* lstm_bhh = (const float*)d_in[19];
  const float* lin1_w   = (const float*)d_in[20];
  const float* lin1_b   = (const float*)d_in[21];
  const float* lin3_w   = (const float*)d_in[22];
  const float* lin3_b   = (const float*)d_in[23];
  float* outp = (float*)d_out;

  const int* srcI = eidx;
  const int* dstI = eidx + EE;

  // ---- carve workspace ----
  char* p = (char*)d_ws;
  size_t off = 0;
  auto carve = [&](size_t bytes) -> char* {
    off = alignUp(off, 256);
    char* r = p + off;
    off += bytes;
    return r;
  };
  unsigned short* g     = (unsigned short*)carve((size_t)NN * GW * 2);
  unsigned short* h1    = (unsigned short*)carve((size_t)EE * DIM * 2);
  unsigned short* hb16  = (unsigned short*)carve((size_t)NN * DIM * 2);
  float* hf32           = (float*)carve((size_t)NN * DIM * 4);
  float* agg            = (float*)carve((size_t)NN * DIM * 4);
  float* deg            = (float*)carve((size_t)NN * 4);
  float* invdeg         = (float*)carve((size_t)NN * 4);
  unsigned short* B2sw  = (unsigned short*)carve((size_t)NT_G * 1024 * 2);
  unsigned short* Broot = (unsigned short*)carve((size_t)4  * 1024 * 2);
  unsigned short* Bwih  = (unsigned short*)carve((size_t)12 * 1024 * 2);
  unsigned short* Bwhh  = (unsigned short*)carve((size_t)12 * 1024 * 2);
  float* evec           = (float*)carve((size_t)NN * 4);
  float* qstar          = (float*)carve(2 * DIM * 4);
  float* hl             = (float*)carve(DIM * 4);
  float* cl             = (float*)carve(DIM * 4);
  float* qv             = (float*)carve(DIM * 4);
  float* rvec           = (float*)carve(DIM * 4);
  float* attnmax        = (float*)carve(4);
  float* attnden        = (float*)carve(4);
  if (off > ws_size) return;   // insufficient scratch; nothing safe to do

  // ---- setup ----
  hipMemsetAsync(deg, 0, (size_t)NN * 4, stream);
  k_prep<<<(NT_G + 4 + 12 + 12) * 4, 256, 0, stream>>>(
      em2_w, em2_b, root_w, gru_wih, gru_whh, B2sw, Broot, Bwih, Bwhh);
  k_deg<<<(EE + 255) / 256, 256, 0, stream>>>(dstI, deg);
  k_lin0<<<(NN * DIM) / 256, 256, 0, stream>>>(x, lin0_w, lin0_b, deg, hf32, hb16, invdeg);
  k_em1<<<(EE * DIM) / 256, 256, 0, stream>>>(ea, em1_w, em1_b, h1);

  // ---- 6 message-passing + GRU iterations ----
  for (int it = 0; it < 6; ++it) {
    hipMemsetAsync(agg, 0, (size_t)NN * DIM * 4, stream);
    k_g<<<(MT * NGRP + 7) / 8, 256, 0, stream>>>(hb16, B2sw, g);
    k_msg<<<EE / 4, 256, 0, stream>>>(h1, g, srcI, dstI, agg);
    k_gru<<<(MT + 7) / 8, 256, 0, stream>>>(hb16, hf32, agg, invdeg, conv_b,
                                            Broot, Bwih, Bwhh, gru_bih, gru_bhh);
  }

  // ---- Set2Set ----
  hipMemsetAsync(qstar, 0, 2 * DIM * 4, stream);
  hipMemsetAsync(hl, 0, DIM * 4, stream);
  hipMemsetAsync(cl, 0, DIM * 4, stream);
  for (int it = 0; it < 6; ++it) {
    k_lstm<<<1, 256, 0, stream>>>(qstar, hl, cl, qv, lstm_wih, lstm_whh,
                                  lstm_bih, lstm_bhh, attnmax, attnden, rvec);
    k_attn_logits<<<(NN + 255) / 256, 256, 0, stream>>>(hf32, qv, evec, attnmax);
    k_attn_den<<<(NN + 255) / 256, 256, 0, stream>>>(evec, attnmax, attnden);
    k_attn_r<<<(NN + 63) / 64, 256, 0, stream>>>(evec, hf32, attnden, rvec);
    k_qstar<<<1, 128, 0, stream>>>(qv, rvec, qstar);
  }
  k_final<<<1, 128, 0, stream>>>(qstar, lin1_w, lin1_b, lin3_w, lin3_b, outp);
}
